// SudokuGNN_42082089566585
// MI455X (gfx1250) — matrix operand order repaired and checked by reading, not compiled, not run on hardware
//
#include <hip/hip_runtime.h>

// ---------------------------------------------------------------------------
// SudokuGNN: 3-layer GAT on MI455X (gfx1250, wave32).
// GEMMs: v_wmma_f32_16x16x32_bf16, fragments loaded straight from global
//        memory with b128 loads (weights pre-transposed to K-major bf16),
//        no LDS staging, no barriers -> fully pipelineable K loop.
// Edge softmax + aggregation: fp32 hardware atomics (memory-bound part).
// ---------------------------------------------------------------------------

#define NN 131072
#define EE 1048576
#define ET (EE + NN)   // edges + self loops = 1179648
#define HID 64

typedef unsigned short ushort_t;
typedef unsigned int   uint_t;

typedef __bf16 bf16_t;
typedef __attribute__((ext_vector_type(16))) bf16_t v16bf;
typedef __attribute__((ext_vector_type(8)))  float  v8f;

union FragBF { v16bf v; uint_t u[8]; uint4 q[2]; };

// ---- helpers --------------------------------------------------------------

__device__ __forceinline__ ushort_t f2bf(float x) {
    uint_t u = __float_as_uint(x);
    u += 0x7FFFu + ((u >> 16) & 1u);      // round-to-nearest-even
    return (ushort_t)(u >> 16);
}

// order-preserving float <-> uint encoding (for atomicMax on floats)
__device__ __forceinline__ uint_t fenc(float f) {
    uint_t u = __float_as_uint(f);
    return (u & 0x80000000u) ? ~u : (u | 0x80000000u);
}
__device__ __forceinline__ float fdec(uint_t e) {
    uint_t u = (e & 0x80000000u) ? (e ^ 0x80000000u) : ~e;
    return __uint_as_float(u);
}
#define ENC_NEG_INF 0x007FFFFFu          // fenc(-inf)

__device__ __forceinline__ float wave_reduce_add(float v) {
    #pragma unroll
    for (int m = 16; m >= 1; m >>= 1) v += __shfl_xor(v, m, 32);
    return v;
}

// ---- 0. embedding lookup -> bf16 activations ------------------------------

__global__ void embed_kernel(const int* __restrict__ x,
                             const float* __restrict__ emb,
                             ushort_t* __restrict__ hbf) {
    int tid = blockIdx.x * blockDim.x + threadIdx.x;   // N*64 threads
    int n = tid >> 6, c = tid & 63;
    hbf[tid] = f2bf(emb[x[n] * HID + c]);
}

// ---- fp32 [Cin,Cout] -> bf16 K-major Wt[Cout][Cin] ------------------------

__global__ void convert_wt_kernel(const float* __restrict__ W,
                                  ushort_t* __restrict__ Wt,
                                  int Cin, int Cout) {
    int tid = blockIdx.x * blockDim.x + threadIdx.x;
    if (tid < Cin * Cout) {
        int n = tid / Cin, k = tid - n * Cin;
        Wt[tid] = f2bf(W[k * Cout + n]);
    }
}

// ---- WMMA GEMM: D[M,Cout] = A[M,CIN] @ W[CIN,Cout] ------------------------
// A: bf16 row-major [M,CIN]; Wt: bf16 K-major [Cout][CIN].
// Block: 256 threads = 8 waves; each wave owns a 16-row strip and computes
// 16x64 of D with 4 accumulators (A fragment reused 4x per K step).
// WMMA fragments are loaded directly from global memory as b128:
//   A lane frag: A[m][kb+hi*16 .. +15]       (32 contiguous bytes)
//   B lane frag: Wt[n][kb+hi*16 .. +15]      (32 contiguous bytes)
// contiguous k = 16*hi + s in both A and B (same order for both, not the ISA table order).

template<int CIN>
__global__ void gat_gemm_bf16(const ushort_t* __restrict__ A,
                              const ushort_t* __restrict__ Wt,
                              float* __restrict__ D, int Cout) {
    const int lane = threadIdx.x & 31;
    const int w    = threadIdx.x >> 5;                  // 0..7
    const int hi   = lane >> 4;                         // 0/1 half-wave
    const int l15  = lane & 15;
    const int rowBase = (blockIdx.y * 8 + w) * 16;      // wave-private rows
    const int colBase = blockIdx.x * 64;

    v8f acc0 = {}, acc1 = {}, acc2 = {}, acc3 = {};

    const ushort_t* Arow = A + (rowBase + l15) * CIN;
    const ushort_t* B0 = Wt + (colBase + 0 * 16 + l15) * CIN + hi * 16;
    const ushort_t* B1 = Wt + (colBase + 1 * 16 + l15) * CIN + hi * 16;
    const ushort_t* B2 = Wt + (colBase + 2 * 16 + l15) * CIN + hi * 16;
    const ushort_t* B3 = Wt + (colBase + 3 * 16 + l15) * CIN + hi * 16;

    #pragma unroll
    for (int kb = 0; kb < CIN; kb += 32) {
        if (kb + 32 < CIN)
            __builtin_prefetch(Arow + kb + 32, 0, 1);   // -> global_prefetch_b8

        FragBF a;
        a.q[0] = *(const uint4*)(Arow + kb + hi * 16);
        a.q[1] = *(const uint4*)(Arow + kb + hi * 16 + 8);

        FragBF b0, b1, b2, b3;
        b0.q[0] = *(const uint4*)(B0 + kb);
        b0.q[1] = *(const uint4*)(B0 + kb + 8);
        b1.q[0] = *(const uint4*)(B1 + kb);
        b1.q[1] = *(const uint4*)(B1 + kb + 8);
        b2.q[0] = *(const uint4*)(B2 + kb);
        b2.q[1] = *(const uint4*)(B2 + kb + 8);
        b3.q[0] = *(const uint4*)(B3 + kb);
        b3.q[1] = *(const uint4*)(B3 + kb + 8);

        acc0 = __builtin_amdgcn_wmma_f32_16x16x32_bf16(false, a.v, false, b0.v,
                                                       (short)0, acc0, false, false);
        acc1 = __builtin_amdgcn_wmma_f32_16x16x32_bf16(false, a.v, false, b1.v,
                                                       (short)0, acc1, false, false);
        acc2 = __builtin_amdgcn_wmma_f32_16x16x32_bf16(false, a.v, false, b2.v,
                                                       (short)0, acc2, false, false);
        acc3 = __builtin_amdgcn_wmma_f32_16x16x32_bf16(false, a.v, false, b3.v,
                                                       (short)0, acc3, false, false);
    }

    // D fragment: lane = column, rows rowBase + hi*8 + v   (ISA C/D layout)
    #pragma unroll
    for (int v = 0; v < 8; ++v) {
        int m = rowBase + hi * 8 + v;
        float* drow = D + m * Cout + colBase + l15;
        drow[0]  = acc0[v];
        drow[16] = acc1[v];
        drow[32] = acc2[v];
        drow[48] = acc3[v];
    }
}

// ---- attention coefficients: alpha_s/d[n,h] = dot(h[n,h,:], a_{s,d}[h,:]) --
// one wave per (node, head); lanes cover C=64 as float2.

template<int H>
__global__ void alpha_kernel(const float* __restrict__ h,
                             const float* __restrict__ a_src,
                             const float* __restrict__ a_dst,
                             float* __restrict__ as_,
                             float* __restrict__ ad_) {
    int gw   = blockIdx.x * 8 + (threadIdx.x >> 5);   // N*H waves
    int lane = threadIdx.x & 31;
    int n = gw / H, hd = gw % H;
    int base = (n * H + hd) * HID;
    int c0 = lane * 2;
    float2 hv = *(const float2*)(h + base + c0);
    float2 vs = *(const float2*)(a_src + hd * HID + c0);
    float2 vd = *(const float2*)(a_dst + hd * HID + c0);
    float ps = wave_reduce_add(hv.x * vs.x + hv.y * vs.y);
    float pd = wave_reduce_add(hv.x * vd.x + hv.y * vd.y);
    if (lane == 0) { as_[n * H + hd] = ps; ad_[n * H + hd] = pd; }
}

// ---- per-layer scratch init -----------------------------------------------

__global__ void init_edge_kernel(float* __restrict__ agg,
                                 uint_t* __restrict__ emax,
                                 float* __restrict__ denom,
                                 int nNH, int nAgg) {
    int tid = blockIdx.x * blockDim.x + threadIdx.x;
    if (tid < nAgg) agg[tid] = 0.0f;
    if (tid < nNH)  { emax[tid] = ENC_NEG_INF; denom[tid] = 0.0f; }
}

// ---- scatter-softmax pass 1: logits + segment max -------------------------

template<int H>
__global__ void edge_pass1(const int* __restrict__ ei,
                           const float* __restrict__ as_,
                           const float* __restrict__ ad_,
                           float* __restrict__ elog,
                           uint_t* __restrict__ emax) {
    int tid = blockIdx.x * blockDim.x + threadIdx.x;   // ET*H threads
    int e = tid / H, hd = tid % H;
    int s, d;
    if (e < EE) { s = ei[e]; d = ei[EE + e]; } else { s = d = e - EE; }
    float lg = as_[s * H + hd] + ad_[d * H + hd];
    lg = (lg >= 0.0f) ? lg : 0.2f * lg;                // leaky_relu
    elog[tid] = lg;
    atomicMax(&emax[d * H + hd], fenc(lg));
}

// ---- pass 2: exp(logit - max), segment sum of denominators ----------------

template<int H>
__global__ void edge_pass2(const int* __restrict__ ei,
                           float* __restrict__ elog,    // in: logit, out: exp
                           const uint_t* __restrict__ emax,
                           float* __restrict__ denom) {
    int tid = blockIdx.x * blockDim.x + threadIdx.x;
    int e = tid / H, hd = tid % H;
    int d = (e < EE) ? ei[EE + e] : (e - EE);
    float m  = fdec(emax[d * H + hd]);
    float ee = __expf(elog[tid] - m);
    elog[tid] = ee;
    unsafeAtomicAdd(&denom[d * H + hd], ee);
}

// ---- pass 3: msg = h[src]*alpha, segment-sum into agg[dst] ----------------
// one wave per edge; lanes parallel over C=64, loop over heads.

template<int H>
__global__ void edge_pass3(const int* __restrict__ ei,
                           const float* __restrict__ h,
                           const float* __restrict__ eexp,
                           const float* __restrict__ denom,
                           float* __restrict__ agg) {
    int e    = blockIdx.x * 8 + (threadIdx.x >> 5);    // ET waves
    int lane = threadIdx.x & 31;
    int s, d;
    if (e < EE) { s = ei[e]; d = ei[EE + e]; } else { s = d = e - EE; }
    #pragma unroll
    for (int hd = 0; hd < H; ++hd) {
        float a = eexp[e * H + hd] / (denom[d * H + hd] + 1e-16f);
        #pragma unroll
        for (int half = 0; half < 2; ++half) {
            int c = lane + 32 * half;
            float msg = h[(s * H + hd) * HID + c] * a;
            unsafeAtomicAdd(&agg[(d * H + hd) * HID + c], msg);
        }
    }
}

// ---- epilogue: +bias, ELU, keep fp32, emit bf16 for next GEMM -------------

template<int HC>
__global__ void epilogue_kernel(float* __restrict__ agg,
                                const float* __restrict__ bias,
                                ushort_t* __restrict__ hbf) {
    int tid = blockIdx.x * blockDim.x + threadIdx.x;   // N*HC threads
    float v = agg[tid] + bias[tid % HC];
    v = (v > 0.0f) ? v : (__expf(v) - 1.0f);           // ELU(alpha=1)
    agg[tid] = v;
    hbf[tid] = f2bf(v);
}

// ---- final FC: out[n,5] = h[n,:] @ fc_w + fc_b ----------------------------

__global__ void fc_kernel(const float* __restrict__ h,
                          const float* __restrict__ fcw,
                          const float* __restrict__ fcb,
                          float* __restrict__ out) {
    int n    = blockIdx.x * 8 + (threadIdx.x >> 5);
    int lane = threadIdx.x & 31;
    float x0 = h[n * HID + lane];
    float x1 = h[n * HID + lane + 32];
    #pragma unroll
    for (int j = 0; j < 5; ++j) {
        float p = x0 * fcw[lane * 5 + j] + x1 * fcw[(lane + 32) * 5 + j];
        p = wave_reduce_add(p);
        if (lane == 0) out[n * 5 + j] = p + fcb[j];
    }
}

// ---------------------------------------------------------------------------

extern "C" void kernel_launch(void* const* d_in, const int* in_sizes, int n_in,
                              void* d_out, int out_size, void* d_ws, size_t ws_size,
                              hipStream_t stream) {
    const int*   x    = (const int*)  d_in[0];
    const int*   ei   = (const int*)  d_in[1];
    const float* emb  = (const float*)d_in[2];
    const float* W1   = (const float*)d_in[3];
    const float* a1s  = (const float*)d_in[4];
    const float* a1d  = (const float*)d_in[5];
    const float* b1   = (const float*)d_in[6];
    const float* W2   = (const float*)d_in[7];
    const float* a2s  = (const float*)d_in[8];
    const float* a2d  = (const float*)d_in[9];
    const float* b2   = (const float*)d_in[10];
    const float* W3   = (const float*)d_in[11];
    const float* a3s  = (const float*)d_in[12];
    const float* a3d  = (const float*)d_in[13];
    const float* b3   = (const float*)d_in[14];
    const float* fcw  = (const float*)d_in[15];
    const float* fcb  = (const float*)d_in[16];
    float* out = (float*)d_out;

    // workspace carve-up (256B aligned)
    char* ws = (char*)d_ws;
    size_t off = 0;
    auto carve = [&](size_t bytes) {
        void* p = ws + off;
        off = (off + bytes + 255) & ~(size_t)255;
        return p;
    };
    ushort_t* hbf   = (ushort_t*)carve((size_t)NN * 256 * 2);  // bf16 activations
    float*    hg    = (float*)   carve((size_t)NN * 256 * 4);  // GEMM output h
    float*    agg   = (float*)   carve((size_t)NN * 256 * 4);  // aggregated out
    ushort_t* wbf   = (ushort_t*)carve(256 * 256 * 2);         // bf16 Wt (K-major)
    float*    as_   = (float*)   carve((size_t)NN * 4 * 4);
    float*    ad_   = (float*)   carve((size_t)NN * 4 * 4);
    uint_t*   emax  = (uint_t*)  carve((size_t)NN * 4 * 4);
    float*    denom = (float*)   carve((size_t)NN * 4 * 4);
    float*    elog  = (float*)   carve((size_t)ET * 4 * 4);    // logits, then exp

    const dim3 blk(256);

    // 0: embedding -> bf16
    embed_kernel<<<(NN * HID) / 256, blk, 0, stream>>>(x, emb, hbf);

    // ---- layer 1: Cin=64, Cout=256, H=4 ----
    convert_wt_kernel<<<(64 * 256) / 256, blk, 0, stream>>>(W1, wbf, 64, 256);
    gat_gemm_bf16<64><<<dim3(256 / 64, NN / 128), blk, 0, stream>>>(hbf, wbf, hg, 256);
    alpha_kernel<4><<<(NN * 4) / 8, blk, 0, stream>>>(hg, a1s, a1d, as_, ad_);
    init_edge_kernel<<<(NN * 256) / 256, blk, 0, stream>>>(agg, emax, denom, NN * 4, NN * 256);
    edge_pass1<4><<<((size_t)ET * 4) / 256, blk, 0, stream>>>(ei, as_, ad_, elog, emax);
    edge_pass2<4><<<((size_t)ET * 4) / 256, blk, 0, stream>>>(ei, elog, emax, denom);
    edge_pass3<4><<<ET / 8, blk, 0, stream>>>(ei, hg, elog, denom, agg);
    epilogue_kernel<256><<<(NN * 256) / 256, blk, 0, stream>>>(agg, b1, hbf);

    // ---- layer 2: Cin=256, Cout=256, H=4 ----
    convert_wt_kernel<<<(256 * 256) / 256, blk, 0, stream>>>(W2, wbf, 256, 256);
    gat_gemm_bf16<256><<<dim3(256 / 64, NN / 128), blk, 0, stream>>>(hbf, wbf, hg, 256);
    alpha_kernel<4><<<(NN * 4) / 8, blk, 0, stream>>>(hg, a2s, a2d, as_, ad_);
    init_edge_kernel<<<(NN * 256) / 256, blk, 0, stream>>>(agg, emax, denom, NN * 4, NN * 256);
    edge_pass1<4><<<((size_t)ET * 4) / 256, blk, 0, stream>>>(ei, as_, ad_, elog, emax);
    edge_pass2<4><<<((size_t)ET * 4) / 256, blk, 0, stream>>>(ei, elog, emax, denom);
    edge_pass3<4><<<ET / 8, blk, 0, stream>>>(ei, hg, elog, denom, agg);
    epilogue_kernel<256><<<(NN * 256) / 256, blk, 0, stream>>>(agg, b2, hbf);

    // ---- layer 3: Cin=256, Cout=64, H=1 (mean over 1 head == identity) ----
    convert_wt_kernel<<<(256 * 64) / 256, blk, 0, stream>>>(W3, wbf, 256, 64);
    gat_gemm_bf16<256><<<dim3(64 / 64, NN / 128), blk, 0, stream>>>(hbf, wbf, hg, 64);
    alpha_kernel<1><<<NN / 8, blk, 0, stream>>>(hg, a3s, a3d, as_, ad_);
    init_edge_kernel<<<(NN * 64) / 256, blk, 0, stream>>>(agg, emax, denom, NN, NN * 64);
    edge_pass1<1><<<ET / 256, blk, 0, stream>>>(ei, as_, ad_, elog, emax);
    edge_pass2<1><<<ET / 256, blk, 0, stream>>>(ei, elog, emax, denom);
    edge_pass3<1><<<ET / 8, blk, 0, stream>>>(ei, hg, elog, denom, agg);
    epilogue_kernel<64><<<(NN * 64) / 256, blk, 0, stream>>>(agg, b3, hbf);

    // ---- final FC head ----
    fc_kernel<<<NN / 8, blk, 0, stream>>>(agg, fcw, fcb, out);
}
